// SpatioTemporalAttention_65395172049027
// MI455X (gfx1250) — matrix operand order repaired and checked by reading, not compiled, not run on hardware
//
#include <hip/hip_runtime.h>
#include <hip/hip_bf16.h>

typedef __bf16 bf16_t;
typedef __attribute__((ext_vector_type(16))) __bf16 v16bf;
typedef __attribute__((ext_vector_type(8)))  __bf16 v8bf;
typedef __attribute__((ext_vector_type(4)))  __bf16 v4bf;
typedef __attribute__((ext_vector_type(8)))  float  v8f;
typedef __attribute__((ext_vector_type(4)))  float  v4f;

#define NB 16
#define LL 4096
#define CC 256
#define HH 8
#define MROWS (NB * LL)           // 65536
#define SCALE_F 0.17677669529663687f

// ---------------------------------------------------------------------------
// f32 -> bf16 convert (vectorized x4)
// ---------------------------------------------------------------------------
__global__ __launch_bounds__(256) void cvt_f32_bf16(const float* __restrict__ in,
                                                    bf16_t* __restrict__ out) {
  int i = (blockIdx.x * 256 + threadIdx.x) * 4;
  v4f x = *(const v4f*)(in + i);
  v4bf y;
  y[0] = (bf16_t)x[0]; y[1] = (bf16_t)x[1];
  y[2] = (bf16_t)x[2]; y[3] = (bf16_t)x[3];
  *(v4bf*)(out + i) = y;
}

// ---------------------------------------------------------------------------
// Weight transpose + convert: W[256(in),256(out)] f32 -> WT[out][in] bf16
// ---------------------------------------------------------------------------
__global__ __launch_bounds__(256) void transpose_cvt(const float* __restrict__ W,
                                                     bf16_t* __restrict__ WT) {
  __shared__ float t[16][17];
  int tx = threadIdx.x & 15, ty = threadIdx.x >> 4;
  int bx = blockIdx.x * 16, by = blockIdx.y * 16;
  t[ty][tx] = W[(by + ty) * CC + bx + tx];
  __syncthreads();
  WT[(bx + ty) * CC + by + tx] = (bf16_t)t[tx][ty];
}

// ---------------------------------------------------------------------------
// GEMM: out[M,256] = A[M,256](bf16) @ WT^T + bias (+resid), WT is [256(out),256(in)]
// Block: 256 threads = 8 waves (2x4), wave tile 32x32 (4 WMMA accumulators).
// Grid: (M/64, 2)
// ---------------------------------------------------------------------------
__global__ __launch_bounds__(256) void gemm_bf16_256(
    const bf16_t* __restrict__ A, const bf16_t* __restrict__ BT,
    const float* __restrict__ bias, const float* __restrict__ resid,
    float* __restrict__ outF, bf16_t* __restrict__ outB) {
  const int lane = threadIdx.x & 31;
  const int w    = threadIdx.x >> 5;
  const int wm   = w >> 2;        // 0..1
  const int wn   = w & 3;         // 0..3
  const int rb   = blockIdx.x * 64 + wm * 32;
  const int cb   = blockIdx.y * 128 + wn * 32;

  // A-fragment layout (16-bit A 16x32): lane m holds K 0-7 & 16-23, lane m+16 holds K 8-15 & 24-31
  const int rA  = rb + (lane & 15);
  const int kAo = (lane >> 4) << 3;   // 0 or 8
  // B-fragment layout (16-bit B 32x16): lanes 0-15 hold K 0-7 & 16-23 of col n, lanes 16-31 hold K 8-15 & 24-31
  const int cB  = cb + (lane & 15);
  const int kBo = (lane >> 4) << 3;   // 0 or 8

  const bf16_t* Arow0 = A + (size_t)rA * CC;
  const bf16_t* Arow1 = Arow0 + 16 * CC;
  const bf16_t* Bcol0 = BT + (size_t)cB * CC;
  const bf16_t* Bcol1 = Bcol0 + 16 * CC;

  v8f acc00 = {}, acc01 = {}, acc10 = {}, acc11 = {};

#pragma unroll
  for (int k = 0; k < CC; k += 32) {
    if (k + 32 < CC) {
      __builtin_prefetch(Arow0 + k + 32 + kAo, 0, 1);
      __builtin_prefetch(Arow1 + k + 32 + kAo, 0, 1);
    }
    v16bf a0, a1, b0, b1;
    ((v8bf*)&a0)[0] = *(const v8bf*)(Arow0 + k + kAo);
    ((v8bf*)&a0)[1] = *(const v8bf*)(Arow0 + k + kAo + 16);
    ((v8bf*)&a1)[0] = *(const v8bf*)(Arow1 + k + kAo);
    ((v8bf*)&a1)[1] = *(const v8bf*)(Arow1 + k + kAo + 16);
    ((v8bf*)&b0)[0] = *(const v8bf*)(Bcol0 + k + kBo);
    ((v8bf*)&b0)[1] = *(const v8bf*)(Bcol0 + k + kBo + 16);
    ((v8bf*)&b1)[0] = *(const v8bf*)(Bcol1 + k + kBo);
    ((v8bf*)&b1)[1] = *(const v8bf*)(Bcol1 + k + kBo + 16);

    acc00 = __builtin_amdgcn_wmma_f32_16x16x32_bf16(false, a0, false, b0, (short)0, acc00, false, false);
    acc01 = __builtin_amdgcn_wmma_f32_16x16x32_bf16(false, a0, false, b1, (short)0, acc01, false, false);
    acc10 = __builtin_amdgcn_wmma_f32_16x16x32_bf16(false, a1, false, b0, (short)0, acc10, false, false);
    acc11 = __builtin_amdgcn_wmma_f32_16x16x32_bf16(false, a1, false, b1, (short)0, acc11, false, false);
  }

  // C/D layout (32-bit 16x16): lane&15 = column, VGPR r = row r (+8 for lanes 16-31)
  const int col0 = cb + (lane & 15);
  const int row0 = rb + ((lane >> 4) << 3);
  v8f accs[2][2] = {{acc00, acc01}, {acc10, acc11}};
#pragma unroll
  for (int mt = 0; mt < 2; ++mt) {
#pragma unroll
    for (int nt = 0; nt < 2; ++nt) {
      const int col = col0 + nt * 16;
      const float bv = bias[col];
#pragma unroll
      for (int r = 0; r < 8; ++r) {
        const int row = row0 + mt * 16 + r;
        const size_t idx = (size_t)row * CC + col;
        float vv = accs[mt][nt][r] + bv;
        if (resid) vv += resid[idx];
        if (outF) outF[idx] = vv;
        if (outB) outB[idx] = (bf16_t)vv;
      }
    }
  }
}

// ---------------------------------------------------------------------------
// Per-row head scores: s[row,h] = (dot_c((x[row,c]*g[n,c]) * Wa[c,h]) + ba[h]) * SCALE
// One wave32 per row; 8 waves per block.
// ---------------------------------------------------------------------------
__global__ __launch_bounds__(256) void head_score(
    const float* __restrict__ x, const float* __restrict__ gate,
    const float* __restrict__ Wa, const float* __restrict__ ba,
    float* __restrict__ s) {
  const int lane = threadIdx.x & 31;
  const int w    = threadIdx.x >> 5;
  const int row  = blockIdx.x * 8 + w;
  const int n    = row >> 12;   // row / L
  const float* xr = x + (size_t)row * CC;
  const float* gr = gate ? (gate + n * CC) : nullptr;
  float p[HH] = {0.f, 0.f, 0.f, 0.f, 0.f, 0.f, 0.f, 0.f};
#pragma unroll
  for (int j = 0; j < 8; ++j) {
    const int c = lane * 8 + j;
    float xv = xr[c];
    if (gr) xv *= gr[c];
    const float* wr = Wa + c * HH;
#pragma unroll
    for (int h = 0; h < HH; ++h) p[h] = fmaf(xv, wr[h], p[h]);
  }
#pragma unroll
  for (int off = 16; off > 0; off >>= 1) {
#pragma unroll
    for (int h = 0; h < HH; ++h) p[h] += __shfl_xor(p[h], off);
  }
  if (lane == 0) {
#pragma unroll
    for (int h = 0; h < HH; ++h)
      s[(size_t)row * HH + h] = (p[h] + ba[h]) * SCALE_F;
  }
}

// ---------------------------------------------------------------------------
// Softmax over L + pooled[n,h,d] = sum_l softmax(s[n,:,h])[l] * val[n,l,h*32+d]
// One block per (h, n). Softmax weights cached in LDS (16 KB of 320 KB).
// ---------------------------------------------------------------------------
__global__ __launch_bounds__(256) void softmax_pool(
    const float* __restrict__ score, const float* __restrict__ val,
    float* __restrict__ pooled) {
  const int h = blockIdx.x, n = blockIdx.y;
  const int t = threadIdx.x;
  __shared__ float red[256];
  __shared__ float wbuf[LL];
  __shared__ float sInvZ;

  const float* sp = score + (size_t)n * LL * HH + h;
  float m = -3.4e38f;
  for (int l = t; l < LL; l += 256) m = fmaxf(m, sp[(size_t)l * HH]);
  red[t] = m; __syncthreads();
  for (int s2 = 128; s2 > 0; s2 >>= 1) {
    if (t < s2) red[t] = fmaxf(red[t], red[t + s2]);
    __syncthreads();
  }
  const float mx = red[0];
  __syncthreads();

  float sum = 0.f;
  for (int l = t; l < LL; l += 256) {
    float e = __expf(sp[(size_t)l * HH] - mx);
    wbuf[l] = e;
    sum += e;
  }
  red[t] = sum; __syncthreads();
  for (int s2 = 128; s2 > 0; s2 >>= 1) {
    if (t < s2) red[t] += red[t + s2];
    __syncthreads();
  }
  if (t == 0) sInvZ = 1.0f / red[0];
  __syncthreads();

  const int g = t >> 5, d = t & 31;
  const float* vp = val + (size_t)n * LL * CC + h * 32 + d;
  float acc = 0.f;
  for (int l = g; l < LL; l += 8) acc += wbuf[l] * vp[(size_t)l * CC];
  red[t] = acc; __syncthreads();
  if (t < 32) {
    float a = 0.f;
#pragma unroll
    for (int gg = 0; gg < 8; ++gg) a += red[gg * 32 + t];
    pooled[n * CC + h * 32 + t] = a * sInvZ;
  }
}

// ---------------------------------------------------------------------------
// kv = v * pooled_k (broadcast over L), output bf16 for the transform GEMM
// ---------------------------------------------------------------------------
__global__ __launch_bounds__(256) void gate_cvt(
    const float* __restrict__ v, const float* __restrict__ g,
    bf16_t* __restrict__ out) {
  int i = (blockIdx.x * 256 + threadIdx.x) * 4;
  const int c   = i & (CC - 1);
  const int row = i >> 8;
  const int n   = row >> 12;
  v4f xv = *(const v4f*)(v + i);
  v4f gv = *(const v4f*)(g + n * CC + c);
  v4bf y;
  y[0] = (bf16_t)(xv[0] * gv[0]); y[1] = (bf16_t)(xv[1] * gv[1]);
  y[2] = (bf16_t)(xv[2] * gv[2]); y[3] = (bf16_t)(xv[3] * gv[3]);
  *(v4bf*)(out + i) = y;
}

// ---------------------------------------------------------------------------
extern "C" void kernel_launch(void* const* d_in, const int* in_sizes, int n_in,
                              void* d_out, int out_size, void* d_ws, size_t ws_size,
                              hipStream_t stream) {
  (void)in_sizes; (void)n_in; (void)out_size; (void)ws_size;
  const float* x_q  = (const float*)d_in[0];
  const float* x_kv = (const float*)d_in[1];
  const float* Wq   = (const float*)d_in[2];
  const float* bq   = (const float*)d_in[3];
  const float* Wqa  = (const float*)d_in[4];
  const float* bqa  = (const float*)d_in[5];
  const float* Wk   = (const float*)d_in[6];
  const float* bk   = (const float*)d_in[7];
  const float* Wka  = (const float*)d_in[8];
  const float* bka  = (const float*)d_in[9];
  const float* Wv   = (const float*)d_in[10];
  const float* bv   = (const float*)d_in[11];
  const float* Wt   = (const float*)d_in[12];
  const float* bt   = (const float*)d_in[13];
  const float* Wo   = (const float*)d_in[14];
  const float* bo   = (const float*)d_in[15];
  float* out = (float*)d_out;

  // workspace layout
  char* ws = (char*)d_ws;
  const size_t SZ_BF = (size_t)MROWS * CC * sizeof(bf16_t);  // 32 MB
  const size_t SZ_F  = (size_t)MROWS * CC * sizeof(float);   // 64 MB
  const size_t SZ_SC = (size_t)MROWS * HH * sizeof(float);   // 2 MB
  bf16_t* Aq  = (bf16_t*)(ws);
  bf16_t* Akv = (bf16_t*)(ws + SZ_BF);
  float*  qF  = (float*)(ws + 2 * SZ_BF);
  float*  kF  = (float*)(ws + 2 * SZ_BF + SZ_F);      // later reused for v
  float*  sc  = (float*)(ws + 2 * SZ_BF + 2 * SZ_F);
  float*  pQ  = (float*)(ws + 2 * SZ_BF + 2 * SZ_F + SZ_SC);
  float*  pK  = pQ + NB * CC;
  bf16_t* Wts = (bf16_t*)(pK + NB * CC);
  bf16_t* WqT = Wts;
  bf16_t* WkT = Wts + 1 * CC * CC;
  bf16_t* WvT = Wts + 2 * CC * CC;
  bf16_t* WtT = Wts + 3 * CC * CC;
  bf16_t* WoT = Wts + 4 * CC * CC;
  bf16_t* kvB = Aq;    // x_q bf16 no longer needed after Q GEMM
  bf16_t* tB  = Akv;   // x_kv bf16 no longer needed after V GEMM

  const dim3 blk(256);
  const int cvtGrid = (MROWS * CC) / (4 * 256);     // 16384
  const dim3 tGrid(16, 16);
  const dim3 gemmGrid(MROWS / 64, CC / 128);        // (1024, 2)
  const int scGrid = MROWS / 8;                     // 8192
  const dim3 smGrid(HH, NB);                        // (8, 16)

  // convert activations + weights
  cvt_f32_bf16<<<cvtGrid, blk, 0, stream>>>(x_q,  Aq);
  cvt_f32_bf16<<<cvtGrid, blk, 0, stream>>>(x_kv, Akv);
  transpose_cvt<<<tGrid, blk, 0, stream>>>(Wq, WqT);
  transpose_cvt<<<tGrid, blk, 0, stream>>>(Wk, WkT);
  transpose_cvt<<<tGrid, blk, 0, stream>>>(Wv, WvT);
  transpose_cvt<<<tGrid, blk, 0, stream>>>(Wt, WtT);
  transpose_cvt<<<tGrid, blk, 0, stream>>>(Wo, WoT);

  // q = x_q @ Wq + bq
  gemm_bf16_256<<<gemmGrid, blk, 0, stream>>>(Aq, WqT, bq, nullptr, qF, nullptr);
  // q scores -> softmax over L -> pooled_q
  head_score<<<scGrid, blk, 0, stream>>>(qF, nullptr, Wqa, bqa, sc);
  softmax_pool<<<smGrid, blk, 0, stream>>>(sc, qF, pQ);

  // k = x_kv @ Wk + bk; scores gated by pooled_q; pooled_k
  gemm_bf16_256<<<gemmGrid, blk, 0, stream>>>(Akv, WkT, bk, nullptr, kF, nullptr);
  head_score<<<scGrid, blk, 0, stream>>>(kF, pQ, Wka, bka, sc);
  softmax_pool<<<smGrid, blk, 0, stream>>>(sc, kF, pK);

  // v = x_kv @ Wv + bv (reuse kF); kv = v * pooled_k -> bf16
  gemm_bf16_256<<<gemmGrid, blk, 0, stream>>>(Akv, WvT, bv, nullptr, kF, nullptr);
  gate_cvt<<<cvtGrid, blk, 0, stream>>>(kF, pK, kvB);

  // t = kv @ Wt + bt + q  (residual), emitted directly as bf16
  gemm_bf16_256<<<gemmGrid, blk, 0, stream>>>(kvB, WtT, bt, qF, nullptr, tB);

  // out = t @ Wo + bo
  gemm_bf16_256<<<gemmGrid, blk, 0, stream>>>(tB, WoT, bo, nullptr, out, nullptr);
}
